// ProtoNCE_45749991637604
// MI455X (gfx1250) — matrix operand order repaired and checked by reading, not compiled, not run on hardware
//
#include <hip/hip_runtime.h>
#include <hip/hip_bf16.h>

#define DIM      128
#define C_CLS    512
#define CPC      64          // classes per chunk in segment-sum kernel
#define CHUNKS   8
#define ROWS_PER_SLAB 2048
#define PST      136         // padded LDS stride in halves (272B -> conflict-free)

typedef __attribute__((ext_vector_type(16))) _Float16 v16h;
typedef __attribute__((ext_vector_type(8)))  float    v8f;

union FragH { v16h v; uint4 q[2]; };

// ---------------- kernel 0: zero fp32 scratch ----------------
__global__ void k_zero(float* __restrict__ p, int n) {
    int i = blockIdx.x * 256 + threadIdx.x;
    if (i < n) p[i] = 0.0f;
}

// ---------------- kernel 1: row-normalize + segment sums ----------------
// grid = (slabs, CHUNKS); each block accumulates only classes of its chunk
// in a 32KB LDS fp32 accumulator (ds_add_f32), then flushes with global atomics.
__global__ __launch_bounds__(256) void k_segsum(
    const float* __restrict__ h, const int* __restrict__ labels,
    float* __restrict__ sums, float* __restrict__ counts)
{
    __shared__ float acc[CPC * DIM];   // 32 KB
    __shared__ float cnt[CPC];
    for (int i = threadIdx.x; i < CPC * DIM; i += 256) acc[i] = 0.0f;
    if (threadIdx.x < CPC) cnt[threadIdx.x] = 0.0f;
    __syncthreads();

    const int chunk = blockIdx.y;
    const int slab0 = blockIdx.x * ROWS_PER_SLAB;
    const int wave  = threadIdx.x >> 5;
    const int lane  = threadIdx.x & 31;

    for (int r = slab0 + wave; r < slab0 + ROWS_PER_SLAB; r += 8) {
        int lbl = labels[r];
        if ((lbl >> 6) != chunk) continue;           // 64 classes per chunk
        float4 v = *(const float4*)(h + (size_t)r * DIM + lane * 4);
        float ss = v.x*v.x + v.y*v.y + v.z*v.z + v.w*v.w;
        #pragma unroll
        for (int msk = 1; msk < 32; msk <<= 1) ss += __shfl_xor(ss, msk, 32);
        float inv = 1.0f / fmaxf(sqrtf(ss), 1e-12f);
        int base = (lbl & (CPC - 1)) * DIM + lane * 4;
        atomicAdd(&acc[base + 0], v.x * inv);
        atomicAdd(&acc[base + 1], v.y * inv);
        atomicAdd(&acc[base + 2], v.z * inv);
        atomicAdd(&acc[base + 3], v.w * inv);
        if (lane == 0) atomicAdd(&cnt[lbl & (CPC - 1)], 1.0f);
    }
    __syncthreads();
    const int cbase = chunk * CPC;
    for (int i = threadIdx.x; i < CPC * DIM; i += 256)
        atomicAdd(&sums[cbase * DIM + i], acc[i]);
    if (threadIdx.x < CPC)
        atomicAdd(&counts[cbase + threadIdx.x], cnt[threadIdx.x]);
}

// ---------------- kernel 2: prototypes -> f16, plus target-logit correction ----
// protos_c = l2norm(sums_c / max(cnt_c,1)).  Also folds the entire per-sample
// target-logit sum into the loss accumulator using
//   sum_i logit_{i,l_i} = (1/tau) * sum_c <sums_c, protos_c>
// which removes the label gather from the WMMA hot loop entirely.
__global__ __launch_bounds__(256) void k_protos(
    const float* __restrict__ sums, const float* __restrict__ counts,
    _Float16* __restrict__ protos, float* __restrict__ lossAcc)
{
    const int wave = threadIdx.x >> 5;
    const int lane = threadIdx.x & 31;
    const int c = blockIdx.x * 8 + wave;            // 64 blocks * 8 waves = 512
    float rc = 1.0f / fmaxf(counts[c], 1.0f);
    float4 s4 = *(const float4*)(sums + (size_t)c * DIM + lane * 4);
    float4 v;
    v.x = s4.x * rc; v.y = s4.y * rc; v.z = s4.z * rc; v.w = s4.w * rc;
    float ss = v.x*v.x + v.y*v.y + v.z*v.z + v.w*v.w;
    #pragma unroll
    for (int msk = 1; msk < 32; msk <<= 1) ss += __shfl_xor(ss, msk, 32);
    float inv = 1.0f / fmaxf(sqrtf(ss), 1e-12f);
    union { _Float16 hx[4]; uint2 u; } o;
    o.hx[0] = (_Float16)(v.x * inv);
    o.hx[1] = (_Float16)(v.y * inv);
    o.hx[2] = (_Float16)(v.z * inv);
    o.hx[3] = (_Float16)(v.w * inv);
    *(uint2*)(protos + (size_t)c * DIM + lane * 4) = o.u;

    // <sums_c, protos_c> with the f16-rounded prototype values (matches k_loss B)
    float dot = (float)o.hx[0] * s4.x + (float)o.hx[1] * s4.y
              + (float)o.hx[2] * s4.z + (float)o.hx[3] * s4.w;
    #pragma unroll
    for (int msk = 1; msk < 32; msk <<= 1) dot += __shfl_xor(dot, msk, 32);
    if (lane == 0) atomicAdd(lossAcc, -10.0f * dot);   // -(1/tau)*sum of target logits
}

// ---------------- kernel 3: fused WMMA logits + bounded-softmax log-sum-exp ----
// Protos staged (padded) into 139KB LDS via async copy. Each wave owns PAIRS of
// 16-row tiles (32 rows): two A fragment sets + two accumulator chains share
// every B fragment load (halves LDS traffic per WMMA). Logits are bounded by
// 1/tau = 10, so exp() cannot overflow: no max tracking, raw v_exp_f32.
__global__ __launch_bounds__(256) void k_loss(
    const float* __restrict__ h, const _Float16* __restrict__ protos,
    float* __restrict__ lossAcc, int nRows)
{
    extern __shared__ _Float16 pl[];                 // C_CLS * PST halves
    __shared__ float blkLoss;

    // async staged, padded copy: 512 classes * 16 uint4 (LDS stride 17 uint4)
    {
        const uint4* src = (const uint4*)protos;
        for (int i = threadIdx.x; i < C_CLS * 16; i += 256) {
            int c = i >> 4, p = i & 15;
            unsigned ldsOff = (unsigned)(size_t)&((uint4*)pl)[c * 17 + p];
            const uint4* gp = src + i;
            asm volatile("global_load_async_to_lds_b128 %0, %1, off"
                         :: "v"(ldsOff), "v"(gp) : "memory");
        }
        asm volatile("s_wait_asynccnt 0x0" ::: "memory");
    }
    if (threadIdx.x == 0) blkLoss = 0.0f;
    __syncthreads();

    const int wave = threadIdx.x >> 5;
    const int lane = threadIdx.x & 31;
    const int m = lane & 15;                         // A-row / D-column index
    const int g = lane >> 4;                         // lane group
    const int nPairs = nRows >> 5;                   // 32 rows per wave iteration

    for (int tp = blockIdx.x * 8 + wave; tp < nPairs; tp += gridDim.x * 8) {
        const float* rowp0 = h + (size_t)(tp * 32 + m) * DIM;
        const float* rowp1 = rowp0 + (size_t)16 * DIM;

        // normalize both row sets on the fly (1/tau folded into the scale)
        FragH a0[4], a1[4];
        #pragma unroll
        for (int s = 0; s < 2; ++s) {
            const float* rowp = s ? rowp1 : rowp0;
            FragH* a = s ? a1 : a0;
            float ss = 0.0f;
            #pragma unroll
            for (int k = 0; k < 4; ++k) {
                int kb = 32 * k + g * 8;
                float4 p0 = *(const float4*)(rowp + kb);
                float4 p1 = *(const float4*)(rowp + kb + 4);
                float4 p2 = *(const float4*)(rowp + kb + 16);
                float4 p3 = *(const float4*)(rowp + kb + 20);
                ss += p0.x*p0.x + p0.y*p0.y + p0.z*p0.z + p0.w*p0.w;
                ss += p1.x*p1.x + p1.y*p1.y + p1.z*p1.z + p1.w*p1.w;
                ss += p2.x*p2.x + p2.y*p2.y + p2.z*p2.z + p2.w*p2.w;
                ss += p3.x*p3.x + p3.y*p3.y + p3.z*p3.z + p3.w*p3.w;
            }
            ss += __shfl_xor(ss, 16, 32);            // lanes m and m+16 share row m
            float inv = 10.0f / fmaxf(sqrtf(ss), 1e-12f);
            #pragma unroll
            for (int k = 0; k < 4; ++k) {            // pass 2 is L0-hot
                const float* p = rowp + 32 * k + g * 8;
                #pragma unroll
                for (int j = 0; j < 8; ++j) a[k].v[j]     = (_Float16)(p[j]      * inv);
                #pragma unroll
                for (int j = 0; j < 8; ++j) a[k].v[8 + j] = (_Float16)(p[16 + j] * inv);
            }
        }

        float sm0[8], sm1[8];
        #pragma unroll
        for (int r = 0; r < 8; ++r) { sm0[r] = 0.0f; sm1[r] = 0.0f; }

        for (int ct = 0; ct < (C_CLS / 16); ++ct) {
            // B layout: lane%16 = column (class), lane/16 picks 8-K chunks (k=8g..8g+7, 16+8g..)
            const _Float16* colp = pl + (ct * 16 + m) * PST + g * 8;
            FragH b[4];
            #pragma unroll
            for (int k = 0; k < 4; ++k) {            // one B load feeds two chains
                b[k].q[0] = *(const uint4*)(colp + 32 * k);
                b[k].q[1] = *(const uint4*)(colp + 32 * k + 16);
            }
            v8f acc0 = {}, acc1 = {};
            #pragma unroll
            for (int k = 0; k < 4; ++k) {
                acc0 = __builtin_amdgcn_wmma_f32_16x16x32_f16(
                    false, a0[k].v, false, b[k].v, (short)0, acc0, false, false);
                acc1 = __builtin_amdgcn_wmma_f32_16x16x32_f16(
                    false, a1[k].v, false, b[k].v, (short)0, acc1, false, false);
            }
            #pragma unroll
            for (int r = 0; r < 8; ++r) {
                sm0[r] += __expf(acc0[r]);           // |x| <= 10, raw v_exp_f32
                sm1[r] += __expf(acc1[r]);
            }
        }

        // butterfly-sum sumexp across the 16 lanes of this group
        #pragma unroll
        for (int msk = 1; msk < 16; msk <<= 1) {
            #pragma unroll
            for (int r = 0; r < 8; ++r) {
                sm0[r] += __shfl_xor(sm0[r], msk, 32);
                sm1[r] += __shfl_xor(sm1[r], msk, 32);
            }
        }

        if (m == 0) {                                // lanes 0 (rows 0-7) and 16 (rows 8-15)
            float part = 0.0f;
            #pragma unroll
            for (int r = 0; r < 8; ++r) part += logf(sm0[r]) + logf(sm1[r]);
            atomicAdd(&blkLoss, part);               // ds_add_f32
        }
    }

    __syncthreads();
    if (threadIdx.x == 0) atomicAdd(lossAcc, blkLoss);  // one global atomic / block
}

// ---------------- kernel 4: finalize mean ----------------
__global__ void k_final(const float* __restrict__ lossAcc, float* __restrict__ out, float invN) {
    if (threadIdx.x == 0) out[0] = lossAcc[0] * invN;
}

extern "C" void kernel_launch(void* const* d_in, const int* in_sizes, int n_in,
                              void* d_out, int out_size, void* d_ws, size_t ws_size,
                              hipStream_t stream) {
    const float* h      = (const float*)d_in[0];
    const int*   labels = (const int*)d_in[1];
    float*       out    = (float*)d_out;
    const int    nRows  = in_sizes[0] / DIM;

    // workspace layout (bytes)
    char* ws = (char*)d_ws;
    float*     sums    = (float*)(ws);                               // 65536 f32
    float*     counts  = (float*)(ws + (size_t)C_CLS * DIM * 4);     // 512 f32
    float*     lossAcc = (float*)(ws + (size_t)C_CLS * DIM * 4 + C_CLS * 4); // 1 f32
    _Float16*  protos  = (_Float16*)(ws + 264448);                   // 512*128 f16

    const int nZero = C_CLS * DIM + C_CLS + 1;                       // sums+counts+loss
    k_zero<<<(nZero + 255) / 256, 256, 0, stream>>>((float*)ws, nZero);

    dim3 g1(nRows / ROWS_PER_SLAB, CHUNKS);
    k_segsum<<<g1, 256, 0, stream>>>(h, labels, sums, counts);

    k_protos<<<C_CLS / 8, 256, 0, stream>>>(sums, counts, protos, lossAcc);

    const size_t ldsBytes = (size_t)C_CLS * PST * sizeof(_Float16);  // 139264
    hipFuncSetAttribute((const void*)k_loss,
                        hipFuncAttributeMaxDynamicSharedMemorySize, (int)ldsBytes);
    k_loss<<<512, 256, ldsBytes, stream>>>(h, protos, lossAcc, nRows);

    k_final<<<1, 32, 0, stream>>>(lossAcc, out, 1.0f / (float)nRows);
}
